// ClimateRiskGNN_46797963657510
// MI455X (gfx1250) — matrix operand order repaired and checked by reading, not compiled, not run on hardware
//
#include <hip/hip_runtime.h>
#include <hip/hip_bf16.h>
#include <math.h>

#define N_NODES 100000
#define N_EDGES 800000
#define IN_F 43
#define HID 64
#define KG 32
#define H1 4
#define H3 2
#define NEG_SLOPE 0.2f

typedef __attribute__((ext_vector_type(16))) _Float16 v16h;
typedef __attribute__((ext_vector_type(8)))  _Float16 v8h;
typedef __attribute__((ext_vector_type(8)))  float    v8f;

// ---------------------------------------------------------------- utilities
__device__ __forceinline__ void atomicMaxF(float* addr, float val) {
  // standard sign-split trick: correct total order incl. mixed signs
  if (val >= 0.0f) atomicMax((int*)addr, __float_as_int(val));
  else             atomicMin((unsigned int*)addr, __float_as_uint(val));
}

__global__ void k_fill(float* __restrict__ p, float v, int n) {
  int i = blockIdx.x * blockDim.x + threadIdx.x;
  if (i < n) p[i] = v;
}

// -------------------------------------------------- operand packing (f32->f16)
// Ah[m][64] = A[m][k] zero-padded to 64 columns.
__global__ void k_pack_a(const float* __restrict__ A, _Float16* __restrict__ Ah,
                         int M, int K) {
  long idx = (long)blockIdx.x * blockDim.x + threadIdx.x;
  if (idx >= (long)M * 64) return;
  int k = (int)(idx & 63);
  int m = (int)(idx >> 6);
  Ah[idx] = (k < K) ? (_Float16)A[(size_t)m * K + k] : (_Float16)0.0f;
}

// Bt[c][64] = B[k][c] transposed, zero-padded to 64 rows.
__global__ void k_pack_bt(const float* __restrict__ B, _Float16* __restrict__ Bt,
                          int K, int Cout) {
  int idx = blockIdx.x * blockDim.x + threadIdx.x;
  if (idx >= Cout * 64) return;
  int k = idx & 63;
  int c = idx >> 6;
  Bt[idx] = (k < K) ? (_Float16)B[(size_t)k * Cout + c] : (_Float16)0.0f;
}

// ------------------------------------------------- WMMA GEMM (f16 in / f32 acc)
// C[M x Cout] = Ah[M x 64] @ Bt^T  (+bias) (optional ReLU)
// Ah: padded f16 row-major; Bt: padded f16, B transposed (Cout x 64).
// One wave per 16x16 tile; per 32-chunk: 4x global_load_b128 + 1 v_wmma.
__global__ void k_wmma_gemm(const _Float16* __restrict__ Ah,
                            const _Float16* __restrict__ Bt,
                            const float* __restrict__ bias,
                            float* __restrict__ C,
                            int M, int Cout, int relu) {
  const int ntiles = Cout >> 4;
  const int tiles  = (M >> 4) * ntiles;
  const int wave   = (blockIdx.x << 2) + (threadIdx.x >> 5);
  if (wave >= tiles) return;                    // wave-uniform: EXEC stays full
  const int mt   = wave / ntiles;
  const int nt   = wave - mt * ntiles;
  const int row0 = mt << 4;
  const int col0 = nt << 4;
  const int lane = threadIdx.x & 31;
  const int l16  = lane & 15;
  const int hi   = lane >> 4;

  const _Float16* arow = Ah + (size_t)(row0 + l16) * 64;
  const _Float16* bcol = Bt + (size_t)(col0 + l16) * 64;

  v8f acc = {};
#pragma unroll
  for (int k0 = 0; k0 < 64; k0 += 32) {
    // A fragment (ISA 7.12.2): lane covers k = k0+hi*8..+8 and k0+16+hi*8..+8
    v8h alo = *(const v8h*)(arow + k0 + hi * 8);
    v8h ahi = *(const v8h*)(arow + k0 + 16 + hi * 8);
    // B fragment: lane covers k = k0+hi*8..+8 and k0+16+hi*8..+8 of column col0+l16
    v8h blo = *(const v8h*)(bcol + k0 + hi * 8);
    v8h bhi = *(const v8h*)(bcol + k0 + 16 + hi * 8);
    v16h a = __builtin_shufflevector(alo, ahi, 0, 1, 2, 3, 4, 5, 6, 7,
                                     8, 9, 10, 11, 12, 13, 14, 15);
    v16h b = __builtin_shufflevector(blo, bhi, 0, 1, 2, 3, 4, 5, 6, 7,
                                     8, 9, 10, 11, 12, 13, 14, 15);
    acc = __builtin_amdgcn_wmma_f32_16x16x32_f16(false, a, false, b,
                                                 (short)0, acc, false, false);
  }
  const int   col = col0 + l16;
  const float bv  = bias ? bias[col] : 0.0f;
#pragma unroll
  for (int r = 0; r < 8; ++r) {
    int   m = row0 + hi * 8 + r;
    float v = acc[r] + bv;
    if (relu) v = fmaxf(v, 0.0f);
    C[(size_t)m * Cout + col] = v;
  }
}

// ---------------------------------------------------------------- node kernels
__global__ void k_node_prep(const float* __restrict__ x, int* __restrict__ cls, int n) {
  int i = blockIdx.x * blockDim.x + threadIdx.x;
  if (i >= n) return;
  const float* xr = x + (size_t)i * IN_F + (IN_F - KG);
  int c = 0;
#pragma unroll
  for (int f = 0; f < KG; ++f)
    if (xr[f] > 0.5f) c = f;
  cls[i] = c;
}

// w[h] = sum_c gat1_We[h*64+c] * gat1_aedge[h*64+c]
__global__ void k_edge_coef(const float* __restrict__ We,
                            const float* __restrict__ ae,
                            float* __restrict__ wcoef) {
  __shared__ float sm[H1 * HID];
  int t = threadIdx.x;
  sm[t] = We[t] * ae[t];
  __syncthreads();
  if (t < H1) {
    float s = 0.f;
    for (int c = 0; c < HID; ++c) s += sm[t * HID + c];
    wcoef[t] = s;
  }
}

// per-(node,head) attention coefficients a_s, a_d
__global__ void k_node_ab(const float* __restrict__ h,
                          const float* __restrict__ asrc,
                          const float* __restrict__ adst,
                          float* __restrict__ as_o, float* __restrict__ ad_o,
                          int n, int H) {
  int idx = blockIdx.x * blockDim.x + threadIdx.x;
  if (idx >= n * H) return;
  int nd = idx / H, hh = idx - nd * H;
  const float* hp = h + (size_t)nd * H * HID + hh * HID;
  const float* aw = asrc + hh * HID;
  const float* dw = adst + hh * HID;
  float sa = 0.f, sd = 0.f;
  for (int c = 0; c < HID; ++c) {
    float v = hp[c];
    sa = fmaf(v, aw[c], sa);
    sd = fmaf(v, dw[c], sd);
  }
  as_o[idx] = sa;
  ad_o[idx] = sd;
}

// ---------------------------------------------------------------- edge kernels
__global__ void k_edge_logit(const int* __restrict__ ei,
                             const float* __restrict__ as_,
                             const float* __restrict__ ad_,
                             const int* __restrict__ cls,
                             const float* __restrict__ wcoef,   // null => no edge bias
                             const float* __restrict__ same_bias,
                             float* __restrict__ elog,
                             float* __restrict__ mbuf, int H) {
  int idx = blockIdx.x * blockDim.x + threadIdx.x;
  if (idx >= N_EDGES * H) return;
  int e = idx / H, hh = idx - e * H;
  int s = ei[e], d = ei[N_EDGES + e];
  float v = as_[s * H + hh] + ad_[d * H + hh];
  if (wcoef && cls[s] == cls[d]) v += same_bias[0] * wcoef[hh];
  v = (v > 0.f) ? v : NEG_SLOPE * v;
  elog[idx] = v;
  atomicMaxF(&mbuf[d * H + hh], v);
}

__global__ void k_edge_exp(const int* __restrict__ ei,
                           float* __restrict__ elog,
                           const float* __restrict__ mbuf,
                           float* __restrict__ sbuf, int H) {
  int idx = blockIdx.x * blockDim.x + threadIdx.x;
  if (idx >= N_EDGES * H) return;
  int e = idx / H, hh = idx - e * H;
  int d  = ei[N_EDGES + e];
  float ex = expf(elog[idx] - mbuf[d * H + hh]);
  elog[idx] = ex;
  atomicAdd(&sbuf[d * H + hh], ex);
}

// one thread per (edge, channel); sums over heads, one atomic per channel
__global__ void k_edge_agg(const int* __restrict__ ei,
                           const float* __restrict__ elog,
                           const float* __restrict__ sbuf,
                           const float* __restrict__ hfeat,
                           float* __restrict__ acc, int H) {
  long idx = (long)blockIdx.x * blockDim.x + threadIdx.x;
  if (idx >= (long)N_EDGES * HID) return;
  int e = (int)(idx >> 6);
  int c = (int)(idx & 63);
  int s = ei[e], d = ei[N_EDGES + e];
  float v = 0.f;
  for (int hh = 0; hh < H; ++hh) {
    float alpha = elog[e * H + hh] / (sbuf[d * H + hh] + 1e-16f);
    v = fmaf(alpha, hfeat[(size_t)s * H * HID + hh * HID + c], v);
  }
  atomicAdd(&acc[(size_t)d * HID + c], v);
}

__global__ void k_deg(const int* __restrict__ ei, float* __restrict__ deg) {
  int e = blockIdx.x * blockDim.x + threadIdx.x;
  if (e < N_EDGES) atomicAdd(&deg[ei[N_EDGES + e]], 1.0f);
}

__global__ void k_dinv(const float* __restrict__ deg, float* __restrict__ dinv, int n) {
  int i = blockIdx.x * blockDim.x + threadIdx.x;
  if (i < n) dinv[i] = rsqrtf(deg[i] + 1.0f);
}

__global__ void k_gcn_edge(const int* __restrict__ ei,
                           const float* __restrict__ dinv,
                           const float* __restrict__ hg,
                           float* __restrict__ acc) {
  long idx = (long)blockIdx.x * blockDim.x + threadIdx.x;
  if (idx >= (long)N_EDGES * HID) return;
  int e = (int)(idx >> 6);
  int c = (int)(idx & 63);
  int s = ei[e], d = ei[N_EDGES + e];
  atomicAdd(&acc[(size_t)d * HID + c], dinv[s] * dinv[d] * hg[(size_t)s * HID + c]);
}

// ----------------------------------------------------------- combine / BN
__global__ void k_gat1_combine(const float* __restrict__ acc,
                               const float* __restrict__ gat1_b,
                               const int* __restrict__ cls,
                               const float* __restrict__ priorW,
                               const float* __restrict__ priorB,
                               const float* __restrict__ gate,
                               float* __restrict__ out, int n) {
  long idx = (long)blockIdx.x * blockDim.x + threadIdx.x;
  if (idx >= (long)n * HID) return;
  int c  = (int)(idx & 63);
  int nd = (int)(idx >> 6);
  float gs = 1.f / (1.f + expf(-gate[0]));
  float xg = acc[idx] * (1.0f / H1) + gat1_b[c];
  float pr = priorW[cls[nd] * HID + c] + priorB[c];
  out[idx] = (1.f - gs) * xg + gs * pr;
}

__global__ void k_gat3_combine(const float* __restrict__ acc,
                               const float* __restrict__ b,
                               float* __restrict__ out, int n) {
  long idx = (long)blockIdx.x * blockDim.x + threadIdx.x;
  if (idx >= (long)n * HID) return;
  int c = (int)(idx & 63);
  out[idx] = acc[idx] * (1.0f / H3) + b[c];
}

__global__ void k_gcn_combine(const float* __restrict__ acc,
                              const float* __restrict__ hg,
                              const float* __restrict__ dinv,
                              const float* __restrict__ b,
                              float* __restrict__ out, int n) {
  long idx = (long)blockIdx.x * blockDim.x + threadIdx.x;
  if (idx >= (long)n * HID) return;
  int c  = (int)(idx & 63);
  int nd = (int)(idx >> 6);
  float di = dinv[nd];
  out[idx] = acc[idx] + hg[idx] * di * di + b[c];
}

__global__ void k_bn_stats(const float* __restrict__ h, float* __restrict__ stats, int n) {
  __shared__ float sh[512];
  int   c  = threadIdx.x & 63;
  int   rg = (int)(blockIdx.x * (blockDim.x >> 6)) + (threadIdx.x >> 6);
  int   rs = gridDim.x * (blockDim.x >> 6);
  float s = 0.f, sq = 0.f;
  for (int r = rg; r < n; r += rs) {
    float v = h[(size_t)r * HID + c];
    s += v;
    sq += v * v;
  }
  sh[threadIdx.x]       = s;
  sh[256 + threadIdx.x] = sq;
  __syncthreads();
  if (threadIdx.x < 64) {
    for (int g = 1; g < (int)(blockDim.x >> 6); ++g) {
      s  += sh[g * 64 + c];
      sq += sh[256 + g * 64 + c];
    }
    atomicAdd(&stats[c], s);
    atomicAdd(&stats[64 + c], sq);
  }
}

// y = elu( (h-mu)*rsqrt(var+eps)*g + b ) + skip   (safe in-place: pure elementwise)
__global__ void k_bn_apply(const float* __restrict__ h,
                           const float* __restrict__ stats,
                           const float* __restrict__ g,
                           const float* __restrict__ b,
                           const float* __restrict__ skip,
                           float* __restrict__ out, int n) {
  long idx = (long)blockIdx.x * blockDim.x + threadIdx.x;
  if (idx >= (long)n * HID) return;
  int   c    = (int)(idx & 63);
  float invN = 1.0f / (float)n;
  float mu   = stats[c] * invN;
  float var  = stats[64 + c] * invN - mu * mu;
  float rs   = rsqrtf(var + 1e-5f);
  float y = (h[idx] - mu) * rs * g[c] + b[c];
  y = (y > 0.f) ? y : expm1f(y);
  out[idx] = y + skip[idx];
}

// ---------------------------------------------------------------- final head
__global__ void k_final(const float* __restrict__ m1,   // N x 32 (post-ReLU)
                        const float* __restrict__ W2,   // 32
                        const float* __restrict__ b2,   // 1
                        const float* __restrict__ vuln, // 32
                        const float* __restrict__ vs,   // 1
                        const int* __restrict__ cls,
                        float* __restrict__ out, int n) {
  int nd   = blockIdx.x * 8 + (threadIdx.x >> 5);
  int lane = threadIdx.x & 31;
  float v = 0.f;
  if (nd < n) v = m1[(size_t)nd * 32 + lane] * W2[lane];
#pragma unroll
  for (int o = 16; o > 0; o >>= 1) v += __shfl_xor(v, o, 32);
  if (nd < n && lane == 0) {
    float raw  = v + b2[0];
    float gvs  = 1.f / (1.f + expf(-vs[0]));
    float z    = raw + gvs * vuln[cls[nd]];
    out[nd] = 1.f / (1.f + expf(-z));
  }
}

// ================================================================= launch
extern "C" void kernel_launch(void* const* d_in, const int* in_sizes, int n_in,
                              void* d_out, int out_size, void* d_ws, size_t ws_size,
                              hipStream_t stream) {
  (void)in_sizes; (void)n_in; (void)out_size; (void)ws_size;
  const float* x          = (const float*)d_in[0];
  const int*   ei         = (const int*)  d_in[1];
  const float* gat1_W     = (const float*)d_in[2];
  const float* gat1_asrc  = (const float*)d_in[3];
  const float* gat1_adst  = (const float*)d_in[4];
  const float* gat1_aedge = (const float*)d_in[5];
  const float* gat1_We    = (const float*)d_in[6];
  const float* gat1_b     = (const float*)d_in[7];
  const float* kg_prior_W = (const float*)d_in[8];
  const float* kg_prior_b = (const float*)d_in[9];
  const float* gate       = (const float*)d_in[10];
  const float* same_bias  = (const float*)d_in[11];
  const float* bn1_g      = (const float*)d_in[12];
  const float* bn1_b      = (const float*)d_in[13];
  const float* gcn2_W     = (const float*)d_in[14];
  const float* gcn2_b     = (const float*)d_in[15];
  const float* bn2_g      = (const float*)d_in[16];
  const float* bn2_b      = (const float*)d_in[17];
  const float* gat3_W     = (const float*)d_in[18];
  const float* gat3_asrc  = (const float*)d_in[19];
  const float* gat3_adst  = (const float*)d_in[20];
  const float* gat3_b     = (const float*)d_in[21];
  const float* bn3_g      = (const float*)d_in[22];
  const float* bn3_b      = (const float*)d_in[23];
  const float* gcn4_W     = (const float*)d_in[24];
  const float* gcn4_b     = (const float*)d_in[25];
  const float* bn4_g      = (const float*)d_in[26];
  const float* bn4_b      = (const float*)d_in[27];
  const float* skip_W     = (const float*)d_in[28];
  const float* skip_b     = (const float*)d_in[29];
  const float* mlp_W1     = (const float*)d_in[30];
  const float* mlp_b1     = (const float*)d_in[31];
  const float* mlp_W2     = (const float*)d_in[32];
  const float* mlp_b2     = (const float*)d_in[33];
  const float* vuln       = (const float*)d_in[34];
  const float* vuln_scale = (const float*)d_in[35];
  float* out = (float*)d_out;

  char* w = (char*)d_ws;
  size_t off = 0;
  auto take = [&](size_t bytes) -> char* {
    char* p = w + off;
    off += (bytes + 255) & ~(size_t)255;
    return p;
  };
  int*      kg_cls = (int*)     take((size_t)N_NODES * 4);
  float*    dinv   = (float*)   take((size_t)N_NODES * 4);
  float*    deg    = (float*)   take((size_t)N_NODES * 4);
  float*    big    = (float*)   take((size_t)N_NODES * 256 * 4); // h1 / h3 / mlp hidden
  float*    elog   = (float*)   take((size_t)N_EDGES * H1 * 4);
  float*    accb   = (float*)   take((size_t)N_NODES * HID * 4);
  float*    asb    = (float*)   take((size_t)N_NODES * H1 * 4);
  float*    adb    = (float*)   take((size_t)N_NODES * H1 * 4);
  float*    mb     = (float*)   take((size_t)N_NODES * H1 * 4);
  float*    sb     = (float*)   take((size_t)N_NODES * H1 * 4);
  float*    skipb  = (float*)   take((size_t)N_NODES * HID * 4);
  float*    X1     = (float*)   take((size_t)N_NODES * HID * 4);
  float*    X2     = (float*)   take((size_t)N_NODES * HID * 4);
  float*    hg     = (float*)   take((size_t)N_NODES * HID * 4);
  _Float16* Ah     = (_Float16*)take((size_t)N_NODES * 64 * 2);  // packed A (f16)
  _Float16* Bt     = (_Float16*)take((size_t)256 * 64 * 2);      // packed B^T (f16)
  float*    stats  = (float*)   take(128 * 4);
  float*    wcoef  = (float*)   take(4 * 4);

  const int TPB = 256;
  const int gN  = (N_NODES + TPB - 1) / TPB;
  const int gNH = (N_NODES * HID + TPB - 1) / TPB;
  const int gE  = (N_EDGES + TPB - 1) / TPB;
  const int gEC = (int)(((long)N_EDGES * HID + TPB - 1) / TPB);

  auto packA = [&](const float* A, int K) {
    k_pack_a<<<gNH, TPB, 0, stream>>>(A, Ah, N_NODES, K);
  };
  auto gemm = [&](const float* B, int K, int Cout, const float* bias, float* C,
                  int relu) {
    k_pack_bt<<<(Cout * 64 + TPB - 1) / TPB, TPB, 0, stream>>>(B, Bt, K, Cout);
    int tiles  = (N_NODES / 16) * (Cout / 16);
    int blocks = (tiles + 3) / 4;                 // 4 waves / 128-thread block
    k_wmma_gemm<<<blocks, 128, 0, stream>>>(Ah, Bt, bias, C, N_NODES, Cout, relu);
  };

  // ---- prep
  k_node_prep<<<gN, TPB, 0, stream>>>(x, kg_cls, N_NODES);
  k_edge_coef<<<1, H1 * HID, 0, stream>>>(gat1_We, gat1_aedge, wcoef);
  hipMemsetAsync(deg, 0, (size_t)N_NODES * 4, stream);
  k_deg<<<gE, TPB, 0, stream>>>(ei, deg);
  k_dinv<<<gN, TPB, 0, stream>>>(deg, dinv, N_NODES);

  // ---- stage 1: GAT1 + KG prior gate + skip, BN1+ELU
  packA(x, IN_F);                                        // shared by both GEMMs
  gemm(gat1_W, IN_F, H1 * HID, nullptr, big, 0);         // h1 = x @ gat1_W
  gemm(skip_W, IN_F, HID, skip_b, skipb, 0);             // skip = x @ skip_W + b
  k_node_ab<<<(N_NODES * H1 + TPB - 1) / TPB, TPB, 0, stream>>>(
      big, gat1_asrc, gat1_adst, asb, adb, N_NODES, H1);
  k_fill<<<(N_NODES * H1 + TPB - 1) / TPB, TPB, 0, stream>>>(mb, -1e30f, N_NODES * H1);
  hipMemsetAsync(sb, 0, (size_t)N_NODES * H1 * 4, stream);
  hipMemsetAsync(accb, 0, (size_t)N_NODES * HID * 4, stream);
  k_edge_logit<<<(N_EDGES * H1 + TPB - 1) / TPB, TPB, 0, stream>>>(
      ei, asb, adb, kg_cls, wcoef, same_bias, elog, mb, H1);
  k_edge_exp<<<(N_EDGES * H1 + TPB - 1) / TPB, TPB, 0, stream>>>(ei, elog, mb, sb, H1);
  k_edge_agg<<<gEC, TPB, 0, stream>>>(ei, elog, sb, big, accb, H1);
  k_gat1_combine<<<gNH, TPB, 0, stream>>>(accb, gat1_b, kg_cls, kg_prior_W,
                                          kg_prior_b, gate, X1, N_NODES);
  hipMemsetAsync(stats, 0, 128 * 4, stream);
  k_bn_stats<<<1024, TPB, 0, stream>>>(X1, stats, N_NODES);
  k_bn_apply<<<gNH, TPB, 0, stream>>>(X1, stats, bn1_g, bn1_b, skipb, X2, N_NODES);

  // ---- stage 2: GCN2, BN2+ELU, skip=X2
  packA(X2, HID);
  gemm(gcn2_W, HID, HID, nullptr, hg, 0);
  hipMemsetAsync(accb, 0, (size_t)N_NODES * HID * 4, stream);
  k_gcn_edge<<<gEC, TPB, 0, stream>>>(ei, dinv, hg, accb);
  k_gcn_combine<<<gNH, TPB, 0, stream>>>(accb, hg, dinv, gcn2_b, X1, N_NODES);
  hipMemsetAsync(stats, 0, 128 * 4, stream);
  k_bn_stats<<<1024, TPB, 0, stream>>>(X1, stats, N_NODES);
  k_bn_apply<<<gNH, TPB, 0, stream>>>(X1, stats, bn2_g, bn2_b, X2, X1, N_NODES);

  // ---- stage 3: GAT3 (2 heads), BN3+ELU, skip=X1
  packA(X1, HID);
  gemm(gat3_W, HID, H3 * HID, nullptr, big, 0);          // h3 = X1 @ gat3_W
  k_node_ab<<<(N_NODES * H3 + TPB - 1) / TPB, TPB, 0, stream>>>(
      big, gat3_asrc, gat3_adst, asb, adb, N_NODES, H3);
  k_fill<<<(N_NODES * H3 + TPB - 1) / TPB, TPB, 0, stream>>>(mb, -1e30f, N_NODES * H3);
  hipMemsetAsync(sb, 0, (size_t)N_NODES * H3 * 4, stream);
  hipMemsetAsync(accb, 0, (size_t)N_NODES * HID * 4, stream);
  k_edge_logit<<<(N_EDGES * H3 + TPB - 1) / TPB, TPB, 0, stream>>>(
      ei, asb, adb, kg_cls, nullptr, nullptr, elog, mb, H3);
  k_edge_exp<<<(N_EDGES * H3 + TPB - 1) / TPB, TPB, 0, stream>>>(ei, elog, mb, sb, H3);
  k_edge_agg<<<gEC, TPB, 0, stream>>>(ei, elog, sb, big, accb, H3);
  k_gat3_combine<<<gNH, TPB, 0, stream>>>(accb, gat3_b, X2, N_NODES);
  hipMemsetAsync(stats, 0, 128 * 4, stream);
  k_bn_stats<<<1024, TPB, 0, stream>>>(X2, stats, N_NODES);
  k_bn_apply<<<gNH, TPB, 0, stream>>>(X2, stats, bn3_g, bn3_b, X1, X2, N_NODES);

  // ---- stage 4: GCN4, BN4+ELU, skip=X2
  packA(X2, HID);
  gemm(gcn4_W, HID, HID, nullptr, hg, 0);
  hipMemsetAsync(accb, 0, (size_t)N_NODES * HID * 4, stream);
  k_gcn_edge<<<gEC, TPB, 0, stream>>>(ei, dinv, hg, accb);
  k_gcn_combine<<<gNH, TPB, 0, stream>>>(accb, hg, dinv, gcn4_b, X1, N_NODES);
  hipMemsetAsync(stats, 0, 128 * 4, stream);
  k_bn_stats<<<1024, TPB, 0, stream>>>(X1, stats, N_NODES);
  k_bn_apply<<<gNH, TPB, 0, stream>>>(X1, stats, bn4_g, bn4_b, X2, X1, N_NODES);

  // ---- MLP head (ReLU fused in GEMM epilogue) + vuln prior + sigmoid
  packA(X1, HID);
  gemm(mlp_W1, HID, HID / 2, mlp_b1, big, 1);            // N x 32, post-ReLU
  k_final<<<(N_NODES + 7) / 8, TPB, 0, stream>>>(big, mlp_W2, mlp_b2, vuln,
                                                 vuln_scale, kg_cls, out, N_NODES);
}
